// ContextEncoderModel_86569360818729
// MI455X (gfx1250) — matrix operand order repaired and checked by reading, not compiled, not run on hardware
//
#include <hip/hip_runtime.h>
#include <hip/hip_bf16.h>
#include <math.h>

// ---------------------------------------------------------------------------
// ContextEncoderModel on MI455X (gfx1250), wave32 + WMMA bf16 (f32 accum).
// All operands pre-packed into fragment-native [lane][16xbf16] order.
// LSTM scan: persistent kernel, weights staged in LDS (208KB/block), inner
// loop = b128 A loads (global) + b128 B loads (LDS) + v_wmma.
// ---------------------------------------------------------------------------

typedef __attribute__((ext_vector_type(16))) __bf16 v16bf;
typedef __attribute__((ext_vector_type(8)))  float  v8f;

#define B_   256
#define T_   100
#define E_   300
#define H_   512
#define KXC  10           // x-part K chunks (300 padded to 320)
#define KHC  16           // h-part K chunks (512)
#define KC_  26           // total chunks per step
#define NG_  2048         // 4*H
#define ENC_ 1024
#define FRAG 512          // ushorts per packed 16x32 fragment (32 lanes x 16)
#define LDS_BYTES (KC_ * 8 * FRAG * 2 + 64 * 128 * 4)   // 208KB wts + 32KB zs

__device__ __forceinline__ unsigned short f32_bf16_bits(float f) {
  unsigned u = __builtin_bit_cast(unsigned, f);
  unsigned r = (u + 0x7FFFu + ((u >> 16) & 1u)) >> 16;   // RNE
  return (unsigned short)r;
}
__device__ __forceinline__ float sigm(float x) { return 1.0f / (1.0f + __expf(-x)); }

// Load one packed fragment: lane's 16 bf16 values are contiguous (32B aligned).
__device__ __forceinline__ v16bf load_frag(const unsigned short* p) {
  const int lane = threadIdx.x & 31;
  return *(const v16bf*)(p + lane * 16);
}

__device__ __forceinline__ v8f wmma_bf(v16bf a, v16bf b, v8f c) {
  return __builtin_amdgcn_wmma_f32_16x16x32_bf16(false, a, false, b,
                                                 (short)0, c, false, false);
}

// A-fragment scatter math (ISA 7.12.2 16-bit A 16x32):
//   lane = (row&15) + 16*hi,  hi = (koff>>3)&1,  kk = koff-8*hi,
//   elem = kk<8 ? kk : kk-8
__device__ __forceinline__ int a_scatter(int rowIn16, int koff, int* laneOut) {
  int hi = (koff >> 3) & 1;
  int kk = koff - 8 * hi;
  int e = (kk < 8) ? kk : kk - 8;
  *laneOut = rowIn16 + 16 * hi;
  return e;
}

// ---------------------------------------------------------------------------
// Pack B (weights) fp32 [K][N] -> fragment order [chunk][N/16][lane][16].
//   B layout: lane = col + 16*hi ; koff = (e<8 ? e : e+8) + 8*hi.  K zero-padded.
// ---------------------------------------------------------------------------
__global__ void pack_b_kernel(const float* __restrict__ src,
                              unsigned short* __restrict__ dst,
                              int rowOff, int rowsValid, int chunks, int N) {
  const int nf = N >> 4;
  size_t total = (size_t)chunks * nf * FRAG;
  for (size_t i = (size_t)blockIdx.x * blockDim.x + threadIdx.x; i < total;
       i += (size_t)gridDim.x * blockDim.x) {
    int e    = (int)(i & 15);
    int lane = (int)((i >> 4) & 31);
    int cf   = (int)((i >> 9) % nf);
    int c    = (int)((i >> 9) / nf);
    int hi = lane >> 4;
    int col = cf * 16 + (lane & 15);
    int k = c * 32 + ((e < 8) ? e : e + 8) + 8 * hi;
    float v = (k < rowsValid) ? src[(size_t)(k + rowOff) * N + col] : 0.0f;
    dst[i] = f32_bf16_bits(v);
  }
}

// ---------------------------------------------------------------------------
// Pack x (embeddings) fp32 [B][T][E] -> A-frag order [t][g=16rows][c][lane][16]
//   A layout: kk = e<8 ? e : e+8 ; koff = kk + 8*hi.  K=300 zero-padded to 320.
// ---------------------------------------------------------------------------
__global__ void pack_a_x_kernel(const float* __restrict__ x,
                                unsigned short* __restrict__ dst) {
  size_t total = (size_t)T_ * 16 * KXC * FRAG;
  for (size_t i = (size_t)blockIdx.x * blockDim.x + threadIdx.x; i < total;
       i += (size_t)gridDim.x * blockDim.x) {
    int e    = (int)(i & 15);
    int lane = (int)((i >> 4) & 31);
    int c    = (int)((i >> 9) % KXC);
    int g    = (int)((i >> 9) / KXC % 16);
    int t    = (int)(i / ((size_t)16 * KXC * FRAG));
    int hi = lane >> 4;
    int row = g * 16 + (lane & 15);
    int kk = (e < 8) ? e : e + 8;
    int k = c * 32 + kk + 8 * hi;
    float v = (k < E_) ? x[((size_t)row * T_ + t) * E_ + k] : 0.0f;
    dst[i] = f32_bf16_bits(v);
  }
}

// ---------------------------------------------------------------------------
// Persistent LSTM scan: 128 blocks = 2 sides x (4 mTiles x 16 nTiles).
// Block tile = 64 batch rows x 32 h cols -> 64x128 Z tile (gates i,j,f,o).
// Weights for the block's 8 colFrags live in LDS; h is kept ONLY as packed
// bf16 A-fragments, double-buffered by t parity.
// ---------------------------------------------------------------------------
__global__ __launch_bounds__(256) void lstm_scan_kernel(
    const unsigned short* __restrict__ xPL, const unsigned short* __restrict__ xPR,
    const int* __restrict__ lenL, const int* __restrict__ lenR,
    const unsigned short* __restrict__ WxPL, const unsigned short* __restrict__ WxPR,
    const unsigned short* __restrict__ WhPL, const unsigned short* __restrict__ WhPR,
    const float* __restrict__ bL, const float* __restrict__ bR,
    float* __restrict__ cBuf,             // [2 sides][B][H] fp32
    unsigned short* __restrict__ hPerm,   // [2 parity][2 sides][16g][16c][FRAG]
    unsigned short* __restrict__ ctxP,    // [16g][32c][FRAG]  (concat, packed)
    unsigned int* __restrict__ bar) {     // [2 sides][T]
  const int side = (int)blockIdx.x >> 6;
  const int bid  = (int)blockIdx.x & 63;
  const int mT = bid >> 4;               // 0..3  : 64-row tile
  const int nT = bid & 15;               // 0..15 : 32-h-col tile
  const int tid = threadIdx.x;
  const int wave = tid >> 5, lane = tid & 31;
  const int waveM  = wave & 3;           // 16-row group within tile
  const int waveNg = wave >> 2;          // 0..1  : 4 colFrags each

  const unsigned short* xP  = side ? xPR : xPL;
  const int*            ln  = side ? lenR : lenL;
  const unsigned short* WxP = side ? WxPR : WxPL;
  const unsigned short* WhP = side ? WhPR : WhPL;
  const float* bias = side ? bR : bL;
  float* cs = cBuf + (long)side * B_ * H_;
  unsigned int* mybar = bar + side * T_;

  const int g = mT * 4 + waveM;          // 16-row group (0..15)
  const size_t hSideSz = (size_t)16 * KHC * FRAG;

  extern __shared__ unsigned short smem[];
  unsigned short* wlds = smem;                       // [26][8][FRAG]
  float* zs = (float*)(smem + (size_t)KC_ * 8 * FRAG);  // [64][128]

  // ---- stage this block's weight tiles into LDS (fragment order) ----
  // local colFrag fi (0..7): gate = fi>>1, global cf = gate*32 + nT*2 + (fi&1)
  for (int u = tid; u < KC_ * 8 * 64; u += 256) {
    int f8 = u & 63;                 // 16B unit within fragment
    int fi = (u >> 6) & 7;
    int c  = u >> 9;                 // chunk 0..25
    int cfG = (fi >> 1) * 32 + nT * 2 + (fi & 1);
    const unsigned short* srcp =
        (c < KXC) ? WxP + ((size_t)(c * 128 + cfG)) * FRAG
                  : WhP + ((size_t)((c - KXC) * 128 + cfG)) * FRAG;
    *(uint4*)(wlds + ((size_t)(c * 8 + fi)) * FRAG + f8 * 8) =
        *(const uint4*)(srcp + f8 * 8);
  }

  // ---- preload genuinely loop-invariant scalars into registers ----
  float bIv[8], bJv[8], bFv[8], bOv[8];
  int lnv[8];
#pragma unroll
  for (int e8 = 0; e8 < 8; ++e8) {
    int idx = tid + 256 * e8;
    int m = idx >> 5, w = idx & 31;
    int row = mT * 64 + m, ncol = nT * 32 + w;
    bIv[e8] = bias[ncol];
    bJv[e8] = bias[H_ + ncol];
    bFv[e8] = bias[2 * H_ + ncol] + 1.0f;   // forget bias folded in
    bOv[e8] = bias[3 * H_ + ncol];
    lnv[e8] = ln[row];
  }
  __syncthreads();

  for (int t = 0; t < T_; ++t) {
    if (t > 0) {
      if (tid == 0) {
        while (__hip_atomic_load(&mybar[t - 1], __ATOMIC_ACQUIRE,
                                 __HIP_MEMORY_SCOPE_AGENT) < 64u)
          __builtin_amdgcn_s_sleep(1);
      }
      __syncthreads();
    }
    // opaque clobber: stop loop-invariant hoist/spill of fragments across t
    asm volatile("" ::: "memory");

    const unsigned short* hR = hPerm + ((size_t)((t & 1) * 2 + side)) * hSideSz;
    unsigned short*       hW = hPerm + ((size_t)(((t + 1) & 1) * 2 + side)) * hSideSz;

    v8f acc0 = {0, 0, 0, 0, 0, 0, 0, 0};
    v8f acc1 = acc0, acc2 = acc0, acc3 = acc0;

    // ---- x_t @ Wx : A from global (packed), B from LDS ----
    const unsigned short* xa = xP + (((size_t)t * 16 + g) * KXC) * FRAG;
#pragma unroll 2
    for (int kc = 0; kc < KXC; ++kc) {
      v16bf a = load_frag(xa + (size_t)kc * FRAG);
      const unsigned short* wb = wlds + ((size_t)(kc * 8 + 4 * waveNg)) * FRAG;
      acc0 = wmma_bf(a, load_frag(wb + 0 * FRAG), acc0);
      acc1 = wmma_bf(a, load_frag(wb + 1 * FRAG), acc1);
      acc2 = wmma_bf(a, load_frag(wb + 2 * FRAG), acc2);
      acc3 = wmma_bf(a, load_frag(wb + 3 * FRAG), acc3);
    }
    // ---- h_{t-1} @ Wh ----
    const unsigned short* ha = hR + ((size_t)g * KHC) * FRAG;
#pragma unroll 2
    for (int kc = 0; kc < KHC; ++kc) {
      v16bf a = load_frag(ha + (size_t)kc * FRAG);
      const unsigned short* wb =
          wlds + ((size_t)((KXC + kc) * 8 + 4 * waveNg)) * FRAG;
      acc0 = wmma_bf(a, load_frag(wb + 0 * FRAG), acc0);
      acc1 = wmma_bf(a, load_frag(wb + 1 * FRAG), acc1);
      acc2 = wmma_bf(a, load_frag(wb + 2 * FRAG), acc2);
      acc3 = wmma_bf(a, load_frag(wb + 3 * FRAG), acc3);
    }

    // ---- Z -> LDS (C layout: VGPR r, lane l -> m=r+8*(l>>4), n=l&15) ----
    {
      const int mBase = waveM * 16 + ((lane >> 4) << 3);
      const int nBase = waveNg * 64 + (lane & 15);
#pragma unroll
      for (int r = 0; r < 8; ++r) {
        zs[(mBase + r) * 128 + nBase + 0]  = acc0[r];
        zs[(mBase + r) * 128 + nBase + 16] = acc1[r];
        zs[(mBase + r) * 128 + nBase + 32] = acc2[r];
        zs[(mBase + r) * 128 + nBase + 48] = acc3[r];
      }
    }
    __syncthreads();

    // ---- elementwise gate math: 64x32 h elements, 8 per thread ----
#pragma unroll
    for (int e8 = 0; e8 < 8; ++e8) {
      int idx = tid + 256 * e8;
      int m = idx >> 5, w = idx & 31;
      int row  = mT * 64 + m;
      int ncol = nT * 32 + w;             // h column = next-step K index
      float zi = zs[m * 128 + w]      + bIv[e8];
      float zj = zs[m * 128 + 32 + w] + bJv[e8];
      float zf = zs[m * 128 + 64 + w] + bFv[e8];
      float zo = zs[m * 128 + 96 + w] + bOv[e8];
      long ci = (long)row * H_ + ncol;
      float cold = cs[ci];
      float cnew = cold * sigm(zf) + sigm(zi) * tanhf(zj);
      float hnew = tanhf(cnew) * sigm(zo);
      cs[ci] = cnew;
      unsigned short hb = f32_bf16_bits(hnew);
      {  // scatter hnew into next step's A-fragment layout
        int lane2, e = a_scatter(row & 15, ncol & 31, &lane2);
        int gg = row >> 4, cc = ncol >> 5;
        hW[(((size_t)gg * KHC + cc) << 9) + (lane2 << 4) + e] = hb;
      }
      if (t == lnv[e8] - 1) {             // last output -> packed concat ctx
        int col = side * H_ + ncol;       // 0..1023, K of first MLP layer
        int lane2, e = a_scatter(row & 15, col & 31, &lane2);
        int gg = row >> 4, cc = col >> 5;
        ctxP[(((size_t)gg * 32 + cc) << 9) + (lane2 << 4) + e] = hb;
      }
    }
    __syncthreads();
    if (tid == 0) {
      __threadfence();
      __hip_atomic_fetch_add(&mybar[t], 1u, __ATOMIC_RELEASE,
                             __HIP_MEMORY_SCOPE_AGENT);
    }
  }
}

// ---------------------------------------------------------------------------
// relu(A @ B): A packed bf16 [g][K/32][FRAG], B packed bf16 [K/32][N/16][FRAG].
// Optional fp32 output and/or packed-bf16 output (feeds the next layer).
// Block tile 32x64; 8 waves -> wave&1 = 16-row half, wave>>1 = 16-col frag.
// ---------------------------------------------------------------------------
__global__ __launch_bounds__(256) void gemm_relu_kernel(
    const unsigned short* __restrict__ Ap, const unsigned short* __restrict__ Bp,
    float* __restrict__ outF, unsigned short* __restrict__ outP, int N, int K) {
  const int nTiles = N >> 6;
  const int mTile = (int)blockIdx.x / nTiles;
  const int nTile = (int)blockIdx.x % nTiles;
  const int tid = threadIdx.x, wave = tid >> 5, lane = tid & 31;
  const int waveM = wave & 1, waveNg = wave >> 1;
  const int Kc = K >> 5, Nf = N >> 4;
  const int g  = mTile * 2 + waveM;
  const int cf = nTile * 4 + waveNg;

  const unsigned short* Arow = Ap + ((size_t)g * Kc) * FRAG;
  v8f acc = {0, 0, 0, 0, 0, 0, 0, 0};
  for (int kc = 0; kc < Kc; ++kc) {
    v16bf a = load_frag(Arow + (size_t)kc * FRAG);
    v16bf b = load_frag(Bp + ((size_t)(kc * Nf + cf)) * FRAG);
    acc = wmma_bf(a, b, acc);
  }
  const int row0 = mTile * 32 + waveM * 16 + ((lane >> 4) << 3);
  const int col  = nTile * 64 + waveNg * 16 + (lane & 15);
#pragma unroll
  for (int r = 0; r < 8; ++r) {
    float v = acc[r];
    v = v > 0.0f ? v : 0.0f;
    int row = row0 + r;
    if (outF) outF[(size_t)row * N + col] = v;
    if (outP) {
      int lane2, e = a_scatter(row & 15, col & 31, &lane2);
      int gg = row >> 4, cc = col >> 5;
      outP[(((size_t)gg * (N >> 5) + cc) << 9) + (lane2 << 4) + e] =
          f32_bf16_bits(v);
    }
  }
}

// ---------------------------------------------------------------------------
extern "C" void kernel_launch(void* const* d_in, const int* in_sizes, int n_in,
                              void* d_out, int out_size, void* d_ws,
                              size_t ws_size, hipStream_t stream) {
  (void)in_sizes; (void)n_in; (void)out_size; (void)ws_size;
  const float* xL   = (const float*)d_in[0];
  const int*   lenL = (const int*)d_in[1];
  const float* xR   = (const float*)d_in[2];
  const int*   lenR = (const int*)d_in[3];
  const float* Wl   = (const float*)d_in[4];
  const float* bl   = (const float*)d_in[5];
  const float* Wr   = (const float*)d_in[6];
  const float* br   = (const float*)d_in[7];
  const float* Wt   = (const float*)d_in[8];
  const float* Whid = (const float*)d_in[9];
  float* out = (float*)d_out;

  char* ws = (char*)d_ws;
  size_t off = 0;
  auto take = [&](size_t bytes) {
    char* p = ws + off;
    off += (bytes + 255) & ~(size_t)255;
    return p;
  };
  const size_t fb = (size_t)FRAG * 2;  // bytes per fragment
  unsigned short* WxPL = (unsigned short*)take(KXC * 128 * fb);
  unsigned short* WxPR = (unsigned short*)take(KXC * 128 * fb);
  unsigned short* WhPL = (unsigned short*)take(KHC * 128 * fb);
  unsigned short* WhPR = (unsigned short*)take(KHC * 128 * fb);
  unsigned short* WtP  = (unsigned short*)take(32 * 64 * fb);
  unsigned short* Hw0P = (unsigned short*)take(32 * 64 * fb);
  unsigned short* Hw1P = (unsigned short*)take(32 * 64 * fb);
  unsigned short* xPL  = (unsigned short*)take((size_t)T_ * 16 * KXC * fb);
  unsigned short* xPR  = (unsigned short*)take((size_t)T_ * 16 * KXC * fb);
  unsigned short* ctxP = (unsigned short*)take(16 * 32 * fb);
  unsigned short* a0P  = (unsigned short*)take(16 * 32 * fb);
  unsigned short* a1P  = (unsigned short*)take(16 * 32 * fb);
  // zero-init region: c state + double-buffered packed h + barrier counters
  char* zbase = ws + off;
  float* cBuf = (float*)take((size_t)2 * B_ * H_ * 4);
  unsigned short* hPerm = (unsigned short*)take((size_t)4 * 16 * KHC * fb);
  unsigned int* bar = (unsigned int*)take((size_t)2 * T_ * 4);
  size_t zbytes = (size_t)((ws + off) - zbase);
  hipMemsetAsync(zbase, 0, zbytes, stream);

  // Fragment-order packing (parallel, off the critical path)
  pack_b_kernel<<<1024, 256, 0, stream>>>(Wl, WxPL, 0, E_, KXC, NG_);
  pack_b_kernel<<<1024, 256, 0, stream>>>(Wr, WxPR, 0, E_, KXC, NG_);
  pack_b_kernel<<<1024, 256, 0, stream>>>(Wl, WhPL, E_, H_, KHC, NG_);
  pack_b_kernel<<<1024, 256, 0, stream>>>(Wr, WhPR, E_, H_, KHC, NG_);
  pack_b_kernel<<<1024, 256, 0, stream>>>(Wt, WtP, 0, ENC_, 32, ENC_);
  pack_b_kernel<<<1024, 256, 0, stream>>>(Whid, Hw0P, 0, ENC_, 32, ENC_);
  pack_b_kernel<<<1024, 256, 0, stream>>>(Whid + (size_t)ENC_ * ENC_, Hw1P,
                                          0, ENC_, 32, ENC_);
  pack_a_x_kernel<<<2048, 256, 0, stream>>>(xL, xPL);
  pack_a_x_kernel<<<2048, 256, 0, stream>>>(xR, xPR);

  // Persistent cooperative scan over T=100 for both LSTMs (240KB dynamic LDS)
  lstm_scan_kernel<<<128, 256, LDS_BYTES, stream>>>(
      xPL, xPR, lenL, lenR, WxPL, WxPR, WhPL, WhPR, bl, br, cBuf, hPerm,
      ctxP, bar);

  // MLP head: relu(ctx @ Wt) -> relu(@hw0) -> relu(@hw1) -> fp32 out
  const int gblocks = (B_ / 32) * (ENC_ / 64);
  gemm_relu_kernel<<<gblocks, 256, 0, stream>>>(ctxP, WtP, nullptr, a0P,
                                                ENC_, ENC_);
  gemm_relu_kernel<<<gblocks, 256, 0, stream>>>(a0P, Hw0P, nullptr, a1P,
                                                ENC_, ENC_);
  gemm_relu_kernel<<<gblocks, 256, 0, stream>>>(a1P, Hw1P, out, nullptr,
                                                ENC_, ENC_);
}
